// QuantumKernelMethod_65481071398073
// MI455X (gfx1250) — compile-verified, matrix operand order checked by reading, not run on hardware
//
#include <hip/hip_runtime.h>
#include <hip/hip_bf16.h>
#include <math.h>

// ---------------------------------------------------------------------------
// Anisotropic RBF Gram matrix on MI455X (gfx1250, wave32, WMMA).
//
// K[n,m] = exp(-( sqx[n] + sqy[m] - 2 * sum_d x[n,d] * (g[d]*y[m,d]) ))
//
// Roofline: output = 268 MB fp32 @ 23.3 TB/s => ~11.5 us floor. Inputs are
// ~4 MB total -> fully L2 resident (192 MB), so the cross-term GEMM reads all
// come from L2 and the only HBM traffic is the streamed output. The 8.6 GFLOP
// cross GEMM needs ~750 TFLOP/s to stay under the write-bandwidth bound, so it
// runs on v_wmma_f32_16x16x32_f16 with a two-term f16 hi/lo split (hi*hi +
// hi*lo + lo*hi) giving ~fp32 accuracy at 3x a trivially cheap f16 matmul.
//
// Epilogue is exp2-based: row norms are pre-scaled by log2(e) in the prep
// pass, so the per-element cost is one v_fma + one v_exp_f32 (trans, which
// co-executes with VALU) + one non-temporal b32 store -- the minimum for this
// fusion, important because it runs 67M times on a bandwidth-bound kernel.
//
// All device-side indexing is 32-bit (every buffer offset < 2^31 bytes) so
// loads/stores use the GVS addressing mode (sgpr 64-bit base + 32-bit vgpr
// offset, scale_offset) instead of burning VALU on 64-bit vector adds.
// ---------------------------------------------------------------------------

typedef _Float16 v8h  __attribute__((ext_vector_type(8)));
typedef _Float16 v16h __attribute__((ext_vector_type(16)));
typedef float    v8f  __attribute__((ext_vector_type(8)));

#define DIMS 64          // feature dimension D (fixed by the reference)
#define CHUNKS_PER_ROW 8 // DIMS / 8 halves per 16-byte chunk
#define BT   128         // workgroup output tile is BT x BT

#define LOG2E      1.4426950408889634f
#define TWO_LOG2E  2.8853900817779268f

#if __has_builtin(__builtin_amdgcn_exp2f)
  #define QKM_EXP2(x) __builtin_amdgcn_exp2f(x)   // direct v_exp_f32
#else
  #define QKM_EXP2(x) exp2f(x)
#endif

// ---------------------------------------------------------------------------
// Pass 0: per-row scaled norms  sq2 = log2(e) * sum_d g[d]*v[d]^2  (fp32),
// and f16 hi/lo split of the value fed to the GEMM (x rows plain; y rows
// pre-scaled by gamma). One 64-thread block (2 waves) per row; LDS reduction.
// ---------------------------------------------------------------------------
__global__ __launch_bounds__(DIMS)
void qkm_prep_rows(const float* __restrict__ src,
                   const float* __restrict__ gamma,
                   _Float16* __restrict__ hi,
                   _Float16* __restrict__ lo,
                   float* __restrict__ sq2,
                   int applyGamma)
{
    const int row = blockIdx.x;
    const int d   = threadIdx.x;
    const int idx = row * DIMS + d;   // < 2^19, 32-bit safe

    const float g = gamma[d];
    const float v = src[idx];

    // exact fp32 contribution to the gamma-weighted squared norm
    const float s = g * v * v;

    // value entering the cross GEMM (gamma folded into the y operand only)
    const float w   = applyGamma ? (g * v) : v;
    const _Float16 h = (_Float16)w;          // hi part
    const float    r = w - (float)h;         // residual
    hi[idx] = h;
    lo[idx] = (_Float16)r;

    __shared__ float red[DIMS];
    red[d] = s;
    __syncthreads();
    #pragma unroll
    for (int off = DIMS / 2; off > 0; off >>= 1) {
        if (d < off) red[d] += red[d + off];
        __syncthreads();
    }
    if (d == 0) sq2[row] = LOG2E * red[0];   // pre-scaled for exp2 epilogue
}

// ---------------------------------------------------------------------------
// Per-lane fragment load matching the CDNA5 16-bit A (16x32 MxK) VGPR layout:
//   lane group g = lane>>4, row/col t = lane&15
//   element i (0..15):  K = i + 8*g + (i>=8 ? 8 : 0)
// i.e. two contiguous 8-half (16 B) chunks: [k0+8g, +8) and [k0+16+8g, +8).
// B (32x16 KxN) mirrors this with lane -> N column. Chunk index is a 32-bit
// int so the load lowers to saddr + 32-bit voffset (scale_offset).
// ---------------------------------------------------------------------------
__device__ inline v16h qkm_load_frag(const v8h* __restrict__ base8, int chunk)
{
    const v8h c0 = base8[chunk];
    const v8h c1 = base8[chunk + 2];   // +16 halves
    return __builtin_shufflevector(c0, c1,
                                   0, 1, 2, 3, 4, 5, 6, 7,
                                   8, 9, 10, 11, 12, 13, 14, 15);
}

// ---------------------------------------------------------------------------
// Pass 1: tiled cross-GEMM + fused exp2 epilogue.
// 256 threads = 8 waves (wave32). Workgroup tile 128x128; wave tile 32x64
// (2x4 subtiles of 16x16, 64 accumulator VGPRs). K loop: 2 steps of 32.
// Split product: acc += Ah*Bh + Ah*Bl + Al*Bh  (lo*lo dropped, ~2^-22).
// ---------------------------------------------------------------------------
__global__ __launch_bounds__(256)
void qkm_rbf_wmma(const _Float16* __restrict__ xh_,
                  const _Float16* __restrict__ xl_,
                  const _Float16* __restrict__ yh_,
                  const _Float16* __restrict__ yl_,
                  const float* __restrict__ sqx2,
                  const float* __restrict__ sqy2,
                  float* __restrict__ out,
                  int Mtot)
{
    const v8h* __restrict__ xh = (const v8h*)xh_;
    const v8h* __restrict__ xl = (const v8h*)xl_;
    const v8h* __restrict__ yh = (const v8h*)yh_;
    const v8h* __restrict__ yl = (const v8h*)yl_;

    const int wave = threadIdx.x >> 5;
    const int lane = threadIdx.x & 31;
    const int g    = lane >> 4;   // lane group (selects K half per ISA layout)
    const int t    = lane & 15;   // row (A) / column (B) within the 16-tile

    const int wn = blockIdx.y * BT + (wave >> 1) * 32;  // wave row base (2 subtiles)
    const int wm = blockIdx.x * BT + (wave & 1) * 64;   // wave col base (4 subtiles)

    v8f acc[2][4] = {};

    #pragma unroll
    for (int k0 = 0; k0 < DIMS; k0 += 32) {
        const int kc = (k0 >> 3) + g;   // chunk offset within a row
        v16h Ah[2], Al[2], Bh[4], Bl[4];
        #pragma unroll
        for (int i = 0; i < 2; ++i) {
            const int rc = (wn + 16 * i + t) * CHUNKS_PER_ROW + kc;  // 32-bit
            Ah[i] = qkm_load_frag(xh, rc);
            Al[i] = qkm_load_frag(xl, rc);
        }
        #pragma unroll
        for (int j = 0; j < 4; ++j) {
            const int cc = (wm + 16 * j + t) * CHUNKS_PER_ROW + kc;  // 32-bit
            Bh[j] = qkm_load_frag(yh, cc);
            Bl[j] = qkm_load_frag(yl, cc);
        }
        #pragma unroll
        for (int i = 0; i < 2; ++i) {
            #pragma unroll
            for (int j = 0; j < 4; ++j) {
                acc[i][j] = __builtin_amdgcn_wmma_f32_16x16x32_f16(
                    false, Ah[i], false, Bh[j], (short)0, acc[i][j], false, false);
                acc[i][j] = __builtin_amdgcn_wmma_f32_16x16x32_f16(
                    false, Ah[i], false, Bl[j], (short)0, acc[i][j], false, false);
                acc[i][j] = __builtin_amdgcn_wmma_f32_16x16x32_f16(
                    false, Al[i], false, Bh[j], (short)0, acc[i][j], false, false);
            }
        }
    }

    // Epilogue. C/D layout: VGPR r, lanes 0-15 -> M=r (N=lane); lanes 16-31 ->
    // M=r+8 (N=lane-16).
    //   K = exp(-norm) = exp2( 2*log2e*cross - (sqx2[n] + sqy2[m]) )
    // => one v_fma + one v_exp_f32 + one NT b32 store per element.
    // Output element index < 2^27 -> 32-bit addressing.
    #pragma unroll
    for (int i = 0; i < 2; ++i) {
        const int nb = wn + 16 * i + 8 * g;
        float sx[8];
        #pragma unroll
        for (int r = 0; r < 8; ++r) sx[r] = sqx2[nb + r];
        #pragma unroll
        for (int j = 0; j < 4; ++j) {
            const int m    = wm + 16 * j + t;
            const float sy = sqy2[m];
            const int obase = nb * Mtot + m;   // 32-bit safe (< 2^27)
            #pragma unroll
            for (int r = 0; r < 8; ++r) {
                const float tt   = __builtin_fmaf(TWO_LOG2E, acc[i][j][r],
                                                  -(sx[r] + sy));
                const float kval = QKM_EXP2(tt);
                __builtin_nontemporal_store(kval, out + (obase + r * Mtot));
            }
        }
    }
}

// ---------------------------------------------------------------------------
// Launch: inputs are x (N*D f32), y (M*D f32), gamma (D f32), output N*M f32.
// Workspace: xh|xl|yh|yl (f16, N*D / M*D each) + sqx2|sqy2 (f32) ~= 4.3 MB.
// ---------------------------------------------------------------------------
extern "C" void kernel_launch(void* const* d_in, const int* in_sizes, int n_in,
                              void* d_out, int out_size, void* d_ws, size_t ws_size,
                              hipStream_t stream)
{
    const float* x     = (const float*)d_in[0];
    const float* y     = (const float*)d_in[1];
    const float* gamma = (const float*)d_in[2];

    const int N = in_sizes[0] / DIMS;   // 8192
    const int M = in_sizes[1] / DIMS;   // 8192

    char* ws = (char*)d_ws;
    size_t off = 0;
    _Float16* xh = (_Float16*)(ws + off); off += (size_t)N * DIMS * sizeof(_Float16);
    _Float16* xl = (_Float16*)(ws + off); off += (size_t)N * DIMS * sizeof(_Float16);
    _Float16* yh = (_Float16*)(ws + off); off += (size_t)M * DIMS * sizeof(_Float16);
    _Float16* yl = (_Float16*)(ws + off); off += (size_t)M * DIMS * sizeof(_Float16);
    float*    sqx2 = (float*)(ws + off); off += (size_t)N * sizeof(float);
    float*    sqy2 = (float*)(ws + off); off += (size_t)M * sizeof(float);

    qkm_prep_rows<<<N, DIMS, 0, stream>>>(x, gamma, xh, xl, sqx2, 0);
    qkm_prep_rows<<<M, DIMS, 0, stream>>>(y, gamma, yh, yl, sqy2, 1);

    dim3 grid(M / BT, N / BT);   // (64, 64)
    qkm_rbf_wmma<<<grid, 256, 0, stream>>>(xh, xl, yh, yl, sqx2, sqy2,
                                           (float*)d_out, M);
}
